// MulOt_Rescal_35734127902881
// MI455X (gfx1250) — matrix operand order repaired and checked by reading, not compiled, not run on hardware
//
#include <hip/hip_runtime.h>

// ---------------------------------------------------------------------------
// CDNA5 (gfx1250) types for WMMA
// ---------------------------------------------------------------------------
typedef __attribute__((ext_vector_type(16))) __bf16 v16bf;
typedef __attribute__((ext_vector_type(8)))  float  v8f;

#define WMMA_BF16(A, B, C) \
  __builtin_amdgcn_wmma_f32_16x16x32_bf16(false, (A), false, (B), (short)0, (C), false, false)

#define DIM 256      // embedding dim (K)
#define KCH 8        // K chunks of 32: 256/32

// ---------------------------------------------------------------------------
// Kernel 1: gather embeddings, split fp32 -> bf16 hi/lo, and store directly in
// the CDNA5 WMMA fragment layout so the GEMM kernel does flat 32B/lane loads.
//
// A-fragment (16x32 bf16, ISA 7.12.2):
//   lane l -> row M = l%16
//   vgpr v, half h -> K = (v<4?0:16) + (l<16?0:8) + (v&3)*2 + h
// B-fragment (32x16 bf16):
//   lane l -> col N = l%16
//   vgpr v, half h -> K = (v<4?0:16) + (l<16?0:8) + (v&3)*2 + h
// Storage: fragment (tile t, chunk c): lane l owns 16 bf16 at
//   elem_base = ((t*8 + c)*32 + l)*16 + v*2 + h   (contiguous per lane)
// ---------------------------------------------------------------------------
__global__ __launch_bounds__(256)
void prep_frags(const float* __restrict__ emb1, const float* __restrict__ emb2,
                const long long* __restrict__ idx1, const long long* __restrict__ idx2,
                unsigned* __restrict__ Ahi, unsigned* __restrict__ Alo,
                unsigned* __restrict__ Bhi, unsigned* __restrict__ Blo,
                int nTiles) {
  unsigned cnt = (unsigned)nTiles * 2048u;           // threads per matrix
  unsigned tid = blockIdx.x * 256u + threadIdx.x;
  if (tid >= 2u * cnt) return;
  int isB = tid >= cnt;
  unsigned r = isB ? (tid - cnt) : tid;
  unsigned v = r & 7u;
  unsigned l = (r >> 3) & 31u;
  unsigned c = (r >> 8) & 7u;
  unsigned t = r >> 11;
  unsigned row = t * 16u + (l & 15u);

  const long long* idx = isB ? idx2 : idx1;
  const float*     emb = isB ? emb2 : emb1;
  long long srcRow = idx[row] * (long long)DIM;

  unsigned hiW = 0u, loW = 0u;
#pragma unroll
  for (unsigned h = 0; h < 2; ++h) {
    unsigned k;
    if (isB) k = c * 32u + ((v < 4u) ? 0u : 16u) + ((l < 16u) ? 0u : 8u) + (v & 3u) * 2u + h;
    else     k = c * 32u + ((v < 4u) ? 0u : 16u) + ((l < 16u) ? 0u : 8u) + (v & 3u) * 2u + h;
    float x = emb[srcRow + (long long)k];
    __bf16 hb = (__bf16)x;
    float  hf = (float)hb;
    __bf16 lb = (__bf16)(x - hf);
    unsigned hbits = (unsigned)__builtin_bit_cast(unsigned short, hb);
    unsigned lbits = (unsigned)__builtin_bit_cast(unsigned short, lb);
    hiW |= hbits << (16u * h);
    loW |= lbits << (16u * h);
  }
  unsigned word = ((t * 8u + c) * 32u + l) * 8u + v;  // elem_base / 2
  if (isB) { Bhi[word] = hiW; Blo[word] = loW; }
  else     { Ahi[word] = hiW; Alo[word] = loW; }
}

// ---------------------------------------------------------------------------
// Kernel 2: squared row norms + precomputed P-gather byte offsets.
// One wave per gathered row; 8 waves per block.
// ---------------------------------------------------------------------------
__global__ __launch_bounds__(256)
void prep_meta(const float* __restrict__ emb1, const float* __restrict__ emb2,
               const long long* __restrict__ idx1, const long long* __restrict__ idx2,
               float* __restrict__ sq1, float* __restrict__ sq2,
               unsigned long long* __restrict__ rowBase, unsigned* __restrict__ colOff,
               int N, int V) {
  __shared__ float red[256];
  int wave = threadIdx.x >> 5;
  int lane = threadIdx.x & 31;
  int g = blockIdx.x * 8 + wave;
  int isB = 0, row = 0;
  float s = 0.f;
  if (g < 2 * N) {
    isB = g >= N;
    row = isB ? (g - N) : g;
    long long id = isB ? idx2[row] : idx1[row];
    const float* src = (isB ? emb2 : emb1) + id * (long long)DIM;
#pragma unroll
    for (int k = 0; k < 8; ++k) {
      float x = src[lane * 8 + k];
      s += x * x;
    }
  }
  red[threadIdx.x] = s;
  __syncthreads();
  for (int st = 16; st >= 1; st >>= 1) {
    if (lane < st) red[threadIdx.x] += red[threadIdx.x + st];
    __syncthreads();
  }
  if (lane == 0 && g < 2 * N) {
    float val = red[threadIdx.x];
    if (!isB) {
      sq1[row] = val;
      rowBase[row] = (unsigned long long)idx1[row] * (unsigned long long)(V * 4);
    } else {
      sq2[row] = val;
      colOff[row] = (unsigned)idx2[row] * 4u;
    }
  }
}

// ---------------------------------------------------------------------------
// Kernel 3: fused GEMM (bf16x3 emulated-fp32 WMMA) + P gather + weighted sum.
// Block = 4 waves = 64x64 output region; each wave owns a 32x32 block
// (2x2 WMMA tiles). C/D layout: vgpr r -> (M = r or r+8), N = lane%16.
// ---------------------------------------------------------------------------
__global__ __launch_bounds__(128)
void gemm_cost(const v16bf* __restrict__ Ahi, const v16bf* __restrict__ Alo,
               const v16bf* __restrict__ Bhi, const v16bf* __restrict__ Blo,
               const float* __restrict__ sq1, const float* __restrict__ sq2,
               const unsigned long long* __restrict__ rowBase,
               const unsigned* __restrict__ colOff,
               const char* __restrict__ Pbytes,
               float* __restrict__ partials, int blocksJ) {
  __shared__ float red[128];
  int bi = blockIdx.x / blocksJ;
  int bj = blockIdx.x % blocksJ;
  int wave = threadIdx.x >> 5;
  int lane = threadIdx.x & 31;
  int itile0 = bi * 4 + (wave >> 1) * 2;
  int jtile0 = bj * 4 + (wave & 1) * 2;

  v8f acc00 = {}, acc01 = {}, acc10 = {}, acc11 = {};

  for (int c = 0; c < KCH; ++c) {
    int a0 = (itile0 * KCH + c) * 32 + lane;
    int a1 = ((itile0 + 1) * KCH + c) * 32 + lane;
    int b0 = (jtile0 * KCH + c) * 32 + lane;
    int b1 = ((jtile0 + 1) * KCH + c) * 32 + lane;
    v16bf ah0 = Ahi[a0], al0 = Alo[a0];
    v16bf ah1 = Ahi[a1], al1 = Alo[a1];
    v16bf bh0 = Bhi[b0], bl0 = Blo[b0];
    v16bf bh1 = Bhi[b1], bl1 = Blo[b1];

    // emulated fp32: hi*hi + hi*lo + lo*hi, accumulated in f32
    acc00 = WMMA_BF16(ah0, bh0, acc00);
    acc00 = WMMA_BF16(ah0, bl0, acc00);
    acc00 = WMMA_BF16(al0, bh0, acc00);

    acc01 = WMMA_BF16(ah0, bh1, acc01);
    acc01 = WMMA_BF16(ah0, bl1, acc01);
    acc01 = WMMA_BF16(al0, bh1, acc01);

    acc10 = WMMA_BF16(ah1, bh0, acc10);
    acc10 = WMMA_BF16(ah1, bl0, acc10);
    acc10 = WMMA_BF16(al1, bh0, acc10);

    acc11 = WMMA_BF16(ah1, bh1, acc11);
    acc11 = WMMA_BF16(ah1, bl1, acc11);
    acc11 = WMMA_BF16(al1, bh1, acc11);
  }

  // Epilogue: P[idx1_i, idx2_j] * (sq1_i + sq2_j - 2*cross_ij)
  float local = 0.f;
  int jlo = lane & 15;
  int mhi = (lane < 16) ? 0 : 8;
#pragma unroll
  for (int s = 0; s < 2; ++s) {
    int j = (jtile0 + s) * 16 + jlo;
    float s2 = sq2[j];
    unsigned long long co = (unsigned long long)colOff[j];
#pragma unroll
    for (int r2 = 0; r2 < 2; ++r2) {
      v8f a = (s == 0) ? (r2 == 0 ? acc00 : acc10) : (r2 == 0 ? acc01 : acc11);
      int ib = (itile0 + r2) * 16 + mhi;
#pragma unroll
      for (int v = 0; v < 8; ++v) {
        int i = ib + v;
        float p = *(const float*)(Pbytes + rowBase[i] + co);
        local += p * (sq1[i] + s2 - 2.0f * a[v]);
      }
    }
  }

  red[threadIdx.x] = local;
  __syncthreads();
  for (int st = 64; st >= 1; st >>= 1) {
    if (threadIdx.x < st) red[threadIdx.x] += red[threadIdx.x + st];
    __syncthreads();
  }
  if (threadIdx.x == 0) partials[blockIdx.x] = red[0];
}

// ---------------------------------------------------------------------------
// Kernel 4: deterministic final reduction (fixed-order strided + tree).
// ---------------------------------------------------------------------------
__global__ __launch_bounds__(256)
void reduce_partials(const float* __restrict__ partials, int n, float* __restrict__ out) {
  __shared__ float red[256];
  float s = 0.f;
  for (int i = threadIdx.x; i < n; i += 256) s += partials[i];
  red[threadIdx.x] = s;
  __syncthreads();
  for (int st = 128; st >= 1; st >>= 1) {
    if (threadIdx.x < st) red[threadIdx.x] += red[threadIdx.x + st];
    __syncthreads();
  }
  if (threadIdx.x == 0) out[0] = red[0];
}

// ---------------------------------------------------------------------------
extern "C" void kernel_launch(void* const* d_in, const int* in_sizes, int n_in,
                              void* d_out, int out_size, void* d_ws, size_t ws_size,
                              hipStream_t stream) {
  const float*     emb1 = (const float*)d_in[0];
  const float*     emb2 = (const float*)d_in[1];
  const float*     P    = (const float*)d_in[2];
  const long long* idx1 = (const long long*)d_in[3];
  const long long* idx2 = (const long long*)d_in[4];

  const int N = in_sizes[3];            // 8192
  const int V = in_sizes[0] / DIM;      // 10000
  const int nTiles  = N / 16;           // 512
  const int blocks64 = N / 64;          // 128
  const int gridGemm = blocks64 * blocks64;

  // workspace layout (256B aligned slices)
  char* w = (char*)d_ws;
  size_t off = 0;
  auto take = [&](size_t bytes) -> char* {
    char* p = w + off;
    off = (off + bytes + 255) & ~(size_t)255;
    return p;
  };
  size_t fragBytes = (size_t)N * DIM * 2;             // bf16 per matrix: 4 MB
  unsigned* Ahi = (unsigned*)take(fragBytes);
  unsigned* Alo = (unsigned*)take(fragBytes);
  unsigned* Bhi = (unsigned*)take(fragBytes);
  unsigned* Blo = (unsigned*)take(fragBytes);
  float* sq1 = (float*)take((size_t)N * 4);
  float* sq2 = (float*)take((size_t)N * 4);
  unsigned long long* rowBase = (unsigned long long*)take((size_t)N * 8);
  unsigned* colOff = (unsigned*)take((size_t)N * 4);
  float* partials = (float*)take((size_t)gridGemm * 4);

  // 1) gather + bf16 hi/lo split into WMMA fragment layout
  unsigned prepThreads = 2u * (unsigned)nTiles * 2048u;
  prep_frags<<<(prepThreads + 255u) / 256u, 256, 0, stream>>>(
      emb1, emb2, idx1, idx2, Ahi, Alo, Bhi, Blo, nTiles);

  // 2) squared norms + gather offsets
  prep_meta<<<(2 * N + 7) / 8, 256, 0, stream>>>(
      emb1, emb2, idx1, idx2, sq1, sq2, rowBase, colOff, N, V);

  // 3) fused WMMA GEMM + P-weighted cost
  gemm_cost<<<gridGemm, 128, 0, stream>>>(
      (const v16bf*)Ahi, (const v16bf*)Alo, (const v16bf*)Bhi, (const v16bf*)Blo,
      sq1, sq2, rowBase, colOff, (const char*)P, partials, blocks64);

  // 4) deterministic scalar reduction
  reduce_partials<<<1, 256, 0, stream>>>(partials, gridGemm, (float*)d_out);
}
